// PairwiseRankingLoss_6253472382977
// MI455X (gfx1250) — compile-verified, matrix operand order checked by reading, not run on hardware
//
#include <hip/hip_runtime.h>

#define NROWS 8192
#define DDIM  512
#define BLKM  256      // block tile rows (im)
#define BLKN  128      // block tile cols (s)
#define KSTEP 32       // K per WMMA step
#define KITERS (DDIM / KSTEP)   // 16
#define MARGIN 1.0f

typedef __attribute__((ext_vector_type(16))) _Float16 v16h;
typedef __attribute__((ext_vector_type(8)))  _Float16 v8h;
typedef __attribute__((ext_vector_type(8)))  float    v8f;

__device__ __forceinline__ v16h cat8(v8h lo, v8h hi) {
    return __builtin_shufflevector(lo, hi, 0,1,2,3,4,5,6,7,8,9,10,11,12,13,14,15);
}

__global__ void zero_out_kernel(float* out) { out[0] = 0.0f; }

// One wave32 per row: L2-normalize im & s rows, emit f16 copies, compute diag[r] in f32.
__global__ __launch_bounds__(256) void normalize_kernel(
    const float* __restrict__ im, const float* __restrict__ s,
    _Float16* __restrict__ im_h, _Float16* __restrict__ s_h,
    float* __restrict__ diag)
{
    const int gtid = blockIdx.x * blockDim.x + threadIdx.x;
    const int row  = gtid >> 5;
    const int lane = threadIdx.x & 31;
    if (row >= NROWS) return;

    const float* imr = im + (size_t)row * DDIM;
    const float* sr  = s  + (size_t)row * DDIM;

    float vi[16], vs[16];
    float ssi = 0.0f, sss = 0.0f;
#pragma unroll
    for (int q = 0; q < 16; ++q) {
        float a = imr[lane + 32 * q];
        float b = sr [lane + 32 * q];
        vi[q] = a; vs[q] = b;
        ssi += a * a; sss += b * b;
    }
#pragma unroll
    for (int m = 16; m >= 1; m >>= 1) {
        ssi += __shfl_xor(ssi, m, 32);
        sss += __shfl_xor(sss, m, 32);
    }
    const float invi = 1.0f / sqrtf(ssi);
    const float invs = 1.0f / sqrtf(sss);

    float dd = 0.0f;
#pragma unroll
    for (int q = 0; q < 16; ++q) {
        float a = vi[q] * invi;
        float b = vs[q] * invs;
        im_h[(size_t)row * DDIM + lane + 32 * q] = (_Float16)a;
        s_h [(size_t)row * DDIM + lane + 32 * q] = (_Float16)b;
        dd += a * b;
    }
#pragma unroll
    for (int m = 16; m >= 1; m >>= 1) dd += __shfl_xor(dd, m, 32);
    if (lane == 0) diag[row] = dd;
}

// Fused GEMM + ranking-loss reduction.
// Block = 256 threads = 8 waves as 4x2; block tile 256x128 of scores.
// Each wave computes a 64x64 sub-tile as 4x4 fragments of v_wmma_f32_16x16x32_f16.
// LDS is double-buffered and filled with global_load_async_to_lds_b128 (ASYNCcnt),
// overlapping tile k+1 loads with tile k compute.
__global__ __launch_bounds__(256) void gemm_loss_kernel(
    const _Float16* __restrict__ im_h, const _Float16* __restrict__ s_h,
    const float* __restrict__ diag, float* __restrict__ out)
{
    __shared__ __align__(16) _Float16 Atile[2][BLKM * KSTEP];  // 2 x 16 KB
    __shared__ __align__(16) _Float16 Btile[2][BLKN * KSTEP];  // 2 x  8 KB
    __shared__ __align__(16) float rin[BLKM];                  // MARGIN - diag (rows)
    __shared__ __align__(16) float cin[BLKN];                  // MARGIN - diag (cols)

    const int tid  = threadIdx.x;
    const int lane = tid & 31;
    const int w    = tid >> 5;   // wave 0..7
    const int wm   = w >> 1;     // 0..3 -> 64-row slab
    const int wn   = w & 1;      // 0..1 -> 64-col slab
    const int h    = lane >> 4;  // half-wave select
    const int ml   = lane & 15;  // row/col within fragment

    const int rowBase = blockIdx.y * BLKM;  // i base (im rows)
    const int colBase = blockIdx.x * BLKN;  // j base (s rows)

    rin[tid] = MARGIN - diag[rowBase + tid];
    if (tid < BLKN) cin[tid] = MARGIN - diag[colBase + tid];

    // Async staging of one K-slab into LDS buffer `buf`.
    auto stage = [&](int buf, int k0) {
        const uint32_t ldsA = (uint32_t)(uintptr_t)(&Atile[buf][0]);
        const uint32_t ldsB = (uint32_t)(uintptr_t)(&Btile[buf][0]);
        // A: 256 rows x 64B = 1024 x 16B chunks, 4 per thread.
#pragma unroll
        for (int p = 0; p < 4; ++p) {
            const int chunk = tid + p * 256;
            const int r = chunk >> 2;
            const int c = chunk & 3;
            const uint64_t ga  = (uint64_t)(uintptr_t)(im_h + (size_t)(rowBase + r) * DDIM + k0 + c * 8);
            const uint32_t la  = ldsA + (uint32_t)(r * KSTEP + c * 8) * 2u;
            asm volatile("global_load_async_to_lds_b128 %0, %1, off"
                         :: "v"(la), "v"(ga) : "memory");
        }
        // B: 128 rows x 64B = 512 x 16B chunks, 2 per thread.
#pragma unroll
        for (int p = 0; p < 2; ++p) {
            const int chunk = tid + p * 256;
            const int r = chunk >> 2;
            const int c = chunk & 3;
            const uint64_t gb  = (uint64_t)(uintptr_t)(s_h + (size_t)(colBase + r) * DDIM + k0 + c * 8);
            const uint32_t lb  = ldsB + (uint32_t)(r * KSTEP + c * 8) * 2u;
            asm volatile("global_load_async_to_lds_b128 %0, %1, off"
                         :: "v"(lb), "v"(gb) : "memory");
        }
    };

    const v8f vzero = {0.f,0.f,0.f,0.f,0.f,0.f,0.f,0.f};
    v8f acc[4][4];
#pragma unroll
    for (int f = 0; f < 4; ++f)
#pragma unroll
        for (int g = 0; g < 4; ++g) acc[f][g] = vzero;

    stage(0, 0);

    for (int k = 0; k < KITERS; ++k) {
        const int buf = k & 1;
        asm volatile("s_wait_asynccnt 0x0" ::: "memory");
        __syncthreads();
        if (k + 1 < KITERS) stage(buf ^ 1, (k + 1) * KSTEP);

        // Per-lane fragment gathers matching the CDNA5 16-bit A/B layout:
        // lane (h, ml) holds K in {8h..8h+7} U {16+8h..16+8h+7} of row/col ml.
        const _Float16* At = &Atile[buf][0];
        const _Float16* Bt = &Btile[buf][0];
        v16h af[4], bf[4];
#pragma unroll
        for (int f = 0; f < 4; ++f) {
            const _Float16* base = At + (wm * 64 + f * 16 + ml) * KSTEP + h * 8;
            af[f] = cat8(*(const v8h*)base, *(const v8h*)(base + 16));
        }
#pragma unroll
        for (int g = 0; g < 4; ++g) {
            const _Float16* base = Bt + (wn * 64 + g * 16 + ml) * KSTEP + h * 8;
            bf[g] = cat8(*(const v8h*)base, *(const v8h*)(base + 16));
        }
#pragma unroll
        for (int f = 0; f < 4; ++f)
#pragma unroll
            for (int g = 0; g < 4; ++g)
                acc[f][g] = __builtin_amdgcn_wmma_f32_16x16x32_f16(
                    false, af[f], false, bf[g],
                    (short)0, acc[f][g], false, false);
        __syncthreads();
    }

    // Branchless fused epilogue. C layout: VGPR v -> M = v + 8*h, N = ml.
    float lsum = 0.0f;
#pragma unroll
    for (int f = 0; f < 4; ++f) {
        const int iLoc = wm * 64 + f * 16 + 8 * h;   // + v
        const float4 r0 = *(const float4*)&rin[iLoc];
        const float4 r1 = *(const float4*)&rin[iLoc + 4];
        const float rv[8] = {r0.x, r0.y, r0.z, r0.w, r1.x, r1.y, r1.z, r1.w};
#pragma unroll
        for (int g = 0; g < 4; ++g) {
            const int jLoc = wn * 64 + g * 16 + ml;
            const int   j  = colBase + jLoc;
            const float cj = cin[jLoc];
            const v8f   c  = acc[f][g];
#pragma unroll
            for (int v = 0; v < 8; ++v) {
                const int   i  = rowBase + iLoc + v;
                const float sc = c[v];
                float term = fmaxf(0.0f, cj + sc) + fmaxf(0.0f, rv[v] + sc);
                term = (i == j) ? 0.0f : term;   // v_cndmask, no EXEC branches
                lsum += term;
            }
        }
    }

    // wave32 reduce, then one atomic per wave (pre-scaled by 1/N).
#pragma unroll
    for (int m = 16; m >= 1; m >>= 1) lsum += __shfl_xor(lsum, m, 32);
    if (lane == 0) atomicAdd(out, lsum * (1.0f / (float)NROWS));
}

extern "C" void kernel_launch(void* const* d_in, const int* in_sizes, int n_in,
                              void* d_out, int out_size, void* d_ws, size_t ws_size,
                              hipStream_t stream) {
    const float* im = (const float*)d_in[0];
    const float* s  = (const float*)d_in[1];
    float* out = (float*)d_out;

    char* ws = (char*)d_ws;
    _Float16* im_h = (_Float16*)ws;                                     // 8 MB
    _Float16* s_h  = (_Float16*)(ws + (size_t)NROWS * DDIM * 2);        // 8 MB
    float*    diag = (float*)   (ws + (size_t)NROWS * DDIM * 4);        // 32 KB

    zero_out_kernel<<<1, 1, 0, stream>>>(out);
    normalize_kernel<<<NROWS / 8, 256, 0, stream>>>(im, s, im_h, s_h, diag);

    dim3 grid(NROWS / BLKN, NROWS / BLKM);  // 64 x 32 blocks
    gemm_loss_kernel<<<grid, 256, 0, stream>>>(im_h, s_h, diag, out);
}
